// UnifiedZipAdapterF_37812892074193
// MI455X (gfx1250) — matrix operand order repaired and checked by reading, not compiled, not run on hardware
//
#include <hip/hip_runtime.h>
#include <math.h>

// ---------------------------------------------------------------------------
// DinomalyBlock forward for MI455X (gfx1250, wave32, WMMA bf16 -> f32 accum)
// B=8, N=4096, C=768, H=12, D=64, BN=384
// GEMM tiles staged with the Tensor Data Mover (tensor_load_to_lds, TENSORcnt),
// double-buffered in LDS with hardware row padding.
// ---------------------------------------------------------------------------

#define Bb   8
#define Nn   4096
#define Cc   768
#define Hh   12
#define Dd   64
#define BNd  384
#define MTOT (Bb * Nn)          // 32768 rows

typedef __attribute__((ext_vector_type(16))) __bf16 v16bf;
typedef __attribute__((ext_vector_type(8)))  __bf16 v8bf;
typedef __attribute__((ext_vector_type(8)))  float  v8f;
typedef __attribute__((ext_vector_type(4)))  unsigned int v4u;
typedef __attribute__((ext_vector_type(8)))  int v8i;
typedef __attribute__((ext_vector_type(4)))  int v4i;

#if defined(__has_builtin)
#if __has_builtin(__builtin_amdgcn_tensor_load_to_lds) && \
    __has_builtin(__builtin_amdgcn_s_wait_tensorcnt)
#define HAVE_TDM 1
#endif
#endif
#ifndef HAVE_TDM
#define HAVE_TDM 0
#endif

__device__ __forceinline__ v8f wmma_bf16(v16bf a, v16bf b, v8f c) {
  // D = A(16x32 bf16) * B(32x16 bf16) + C(16x16 f32)
  return __builtin_amdgcn_wmma_f32_16x16x32_bf16(
      /*neg_a=*/false, a, /*neg_b=*/false, b,
      /*c_mod=*/(short)0, c, /*reuse_a=*/false, /*reuse_b=*/false);
}

// A fragment: 16x32 (MxK). lane<16 holds row m=lane, K {kb..kb+7, kb+16..kb+23};
// lane>=16 holds row m=lane-16, K {kb+8..kb+15, kb+24..kb+31}.
__device__ __forceinline__ v16bf load_frag_a(const __bf16* base, int ldm,
                                             int row, int kb, int lane) {
  int r = row + (lane & 15);
  int half = lane >> 4;
  const __bf16* p = base + (size_t)r * ldm + kb + half * 8;
  v8bf lo = *(const v8bf*)p;
  v8bf hi = *(const v8bf*)(p + 16);
  return __builtin_shufflevector(lo, hi, 0,1,2,3,4,5,6,7,8,9,10,11,12,13,14,15);
}

// B fragment: 32x16 (KxN'). lane<16 holds col n'=lane, K {kb..kb+7, kb+16..kb+23};
// lane>=16 holds col n'=lane-16, K {kb+8..kb+15, kb+24..kb+31}. Storage: base[n'][k].
__device__ __forceinline__ v16bf load_frag_b(const __bf16* base, int ldk,
                                             int col, int kb, int lane) {
  int c = col + (lane & 15);
  int half = lane >> 4;
  const __bf16* p = base + (size_t)c * ldk + kb + half * 8;
  v8bf lo = *(const v8bf*)p;
  v8bf hi = *(const v8bf*)(p + 16);
  return __builtin_shufflevector(lo, hi, 0,1,2,3,4,5,6,7,8,9,10,11,12,13,14,15);
}

#if HAVE_TDM
// ---------------------------------------------------------------------------
// TDM: issue one tensor_load_to_lds for a 2D bf16 tile [tile_d1 rows x tile_d0
// elems], row stride stride0 (elements), destination LDS with hardware padding:
// pad 4 DWORDs (16B = 8 bf16) after every 32 DWORDs (128B = one 64-elem row),
// i.e. LDS row pitch 72 bf16 -> matches the fragment loaders.
// D# bitfields per CDNA5 ISA.
// ---------------------------------------------------------------------------
__device__ __forceinline__ void tdm_load_tile_bf16(
    unsigned lds_off, const void* gaddr, unsigned tile_d0, unsigned tile_d1,
    unsigned tensor_d0, unsigned tensor_d1, unsigned stride0) {
  unsigned long long ga = (unsigned long long)gaddr;
  v4u g0;
  g0.x = 1u;                                   // count=1 (valid), no gather
  g0.y = lds_off;                              // lds_addr [63:32]
  g0.z = (unsigned)ga;                         // global_addr [95:64]
  g0.w = (unsigned)((ga >> 32) & 0x1FFFFFFu)   // global_addr [120:96]
       | (2u << 30);                           // type=2 ("image")
  v8i g1;
  g1[0] = (int)((1u << 16)                     // data_size = 1 -> 2 bytes
              | (1u << 20)                     // pad_enable
              | (4u << 22)                     // pad_interval: 32 DWORDs
              | (3u << 25));                   // pad_amount: 4 DWORDs
  g1[1] = (int)((tensor_d0 & 0xFFFFu) << 16);  // tensor_dim0[15:0] @ [63:48]
  g1[2] = (int)(((tensor_d0 >> 16) & 0xFFFFu)  // tensor_dim0[31:16]
              | ((tensor_d1 & 0xFFFFu) << 16));// tensor_dim1[15:0]
  g1[3] = (int)(((tensor_d1 >> 16) & 0xFFFFu)  // tensor_dim1[31:16]
              | ((tile_d0 & 0xFFFFu) << 16));  // tile_dim0 @ [127:112]
  g1[4] = (int)(tile_d1 & 0xFFFFu);            // tile_dim1; tile_dim2=0
  g1[5] = (int)stride0;                        // tensor_dim0_stride[31:0]
  g1[6] = 0;                                   // stride0[47:32]=0; stride1=0
  g1[7] = 0;
  v4i gz = {0, 0, 0, 0};
#if __clang_major__ >= 23
  v8i gz8 = {0, 0, 0, 0, 0, 0, 0, 0};
  __builtin_amdgcn_tensor_load_to_lds(g0, g1, gz, gz, gz8, 0);
#else
  __builtin_amdgcn_tensor_load_to_lds(g0, g1, gz, gz, 0);
#endif
}
#endif  // HAVE_TDM

// ---------------------------------------------------------------------------
// Weight convert + transpose: wt[n][k] = (bf16) w[k][n]
// ---------------------------------------------------------------------------
__global__ __launch_bounds__(256) void transpose_bf16_kernel(
    const float* __restrict__ w, __bf16* __restrict__ wt, int Kdim, int Ndim) {
  int idx = blockIdx.x * 256 + threadIdx.x;
  if (idx >= Kdim * Ndim) return;
  int n = idx / Kdim;
  int k = idx - n * Kdim;
  wt[idx] = (__bf16)w[(size_t)k * Ndim + n];
}

// ---------------------------------------------------------------------------
// LayerNorm over C=768 (one block per row). Optional f32 and/or bf16 outputs.
// ---------------------------------------------------------------------------
__global__ __launch_bounds__(256) void layernorm_kernel(
    const float* __restrict__ x, const float* __restrict__ g,
    const float* __restrict__ bet, float* __restrict__ outF,
    __bf16* __restrict__ outB) {
  __shared__ float s1[256], s2[256];
  int row = blockIdx.x, tid = threadIdx.x;
  const float* xr = x + (size_t)row * Cc;
  float v0 = xr[tid], v1 = xr[tid + 256], v2 = xr[tid + 512];
  s1[tid] = v0 + v1 + v2;
  s2[tid] = v0 * v0 + v1 * v1 + v2 * v2;
  __syncthreads();
  for (int o = 128; o > 0; o >>= 1) {
    if (tid < o) { s1[tid] += s1[tid + o]; s2[tid] += s2[tid + o]; }
    __syncthreads();
  }
  float mu = s1[0] * (1.0f / Cc);
  float var = s2[0] * (1.0f / Cc) - mu * mu;
  float rs = rsqrtf(var + 1e-5f);
  float vv[3] = {v0, v1, v2};
  #pragma unroll
  for (int t = 0; t < 3; ++t) {
    int c = tid + t * 256;
    float y = (vv[t] - mu) * rs * g[c] + bet[c];
    if (outF) outF[(size_t)row * Cc + c] = y;
    if (outB) outB[(size_t)row * Cc + c] = (__bf16)y;
  }
}

// ---------------------------------------------------------------------------
// Generic tiled bf16 GEMM: out = A[MxK] * Wt[NxK]^T + bias, fused epilogues.
// Block tile 128x128, K-stage 64, 8 waves (4x2), wave tile 32x64.
// TDM double-buffered LDS staging; WMMA bf16 compute.
// ---------------------------------------------------------------------------
#define EPI_QKV  0
#define EPI_PROJ 1
#define EPI_DOWN 2
#define EPI_UP   3

template <int EPI>
__global__ __launch_bounds__(256) void gemm_bf16_kernel(
    const __bf16* __restrict__ A, const __bf16* __restrict__ Wt,
    const float* __restrict__ bias, int M, int N, int K,
    __bf16* __restrict__ oQ, __bf16* __restrict__ oKT, __bf16* __restrict__ oVT,
    const float* __restrict__ xin, float* __restrict__ oF,
    __bf16* __restrict__ oB, const float* __restrict__ radd) {
  constexpr int BM = 128, BN = 128, BK = 64, LDT = BK + 8;  // pitch 72 bf16
  __shared__ __bf16 As[2][BM][LDT];
  __shared__ __bf16 Bs[2][BN][LDT];
  int tid = threadIdx.x, lane = tid & 31, wave = tid >> 5;
  int wm = wave & 3, wn = wave >> 2;
  int bm = blockIdx.x * BM, bn = blockIdx.y * BN;

  v8f acc[2][4] = {};
  const int nstages = K / BK;

#if HAVE_TDM
  unsigned ldsA0 = (unsigned)(unsigned long long)(const void*)&As[0][0][0];
  unsigned ldsA1 = (unsigned)(unsigned long long)(const void*)&As[1][0][0];
  unsigned ldsB0 = (unsigned)(unsigned long long)(const void*)&Bs[0][0][0];
  unsigned ldsB1 = (unsigned)(unsigned long long)(const void*)&Bs[1][0][0];
  if (wave == 0) {  // prime stage 0
    tdm_load_tile_bf16(ldsA0, A + (size_t)bm * K, BK, BM, K, BM, K);
    tdm_load_tile_bf16(ldsB0, Wt + (size_t)bn * K, BK, BN, K, BN, K);
  }
  for (int s = 0; s < nstages; ++s) {
    int cur = s & 1;
    if (wave == 0) {
      if (s + 1 < nstages) {  // kick DMA for next stage into other buffer
        int k1 = (s + 1) * BK;
        unsigned la = (s + 1) & 1 ? ldsA1 : ldsA0;
        unsigned lb = (s + 1) & 1 ? ldsB1 : ldsB0;
        tdm_load_tile_bf16(la, A + (size_t)bm * K + k1, BK, BM, K - k1, BM, K);
        tdm_load_tile_bf16(lb, Wt + (size_t)bn * K + k1, BK, BN, K - k1, BN, K);
        __builtin_amdgcn_s_wait_tensorcnt(2);  // stage s complete (in-order)
      } else {
        __builtin_amdgcn_s_wait_tensorcnt(0);
      }
    }
    __syncthreads();
    const __bf16* Ab = &As[cur][0][0];
    const __bf16* Bb_ = &Bs[cur][0][0];
#else
  for (int s = 0; s < nstages; ++s) {
    int k0 = s * BK;
    int c = (tid & 7) * 8;
    #pragma unroll
    for (int it = 0; it < 4; ++it) {
      int r = (tid >> 3) + it * 32;
      *(v8bf*)&As[0][r][c] = *(const v8bf*)(A  + (size_t)(bm + r) * K + k0 + c);
      *(v8bf*)&Bs[0][r][c] = *(const v8bf*)(Wt + (size_t)(bn + r) * K + k0 + c);
    }
    if (k0 + BK < K) {
      __builtin_prefetch(A  + (size_t)(bm + (tid >> 3)) * K + k0 + BK + c, 0, 1);
      __builtin_prefetch(Wt + (size_t)(bn + (tid >> 3)) * K + k0 + BK + c, 0, 1);
    }
    __syncthreads();
    const __bf16* Ab = &As[0][0][0];
    const __bf16* Bb_ = &Bs[0][0][0];
#endif
    #pragma unroll
    for (int kk = 0; kk < BK; kk += 32) {
      v16bf af[2], bfr[4];
      #pragma unroll
      for (int i = 0; i < 2; ++i)
        af[i] = load_frag_a(Ab, LDT, wm * 32 + i * 16, kk, lane);
      #pragma unroll
      for (int j = 0; j < 4; ++j)
        bfr[j] = load_frag_b(Bb_, LDT, wn * 64 + j * 16, kk, lane);
      #pragma unroll
      for (int i = 0; i < 2; ++i)
        #pragma unroll
        for (int j = 0; j < 4; ++j)
          acc[i][j] = wmma_bf16(af[i], bfr[j], acc[i][j]);
    }
    __syncthreads();
  }

  // Epilogue. C/D fragment: elem e -> row m=e+8*(lane>=16), col n'=lane%16.
  int half = lane >> 4, l15 = lane & 15;
  #pragma unroll
  for (int i = 0; i < 2; ++i) {
    #pragma unroll
    for (int j = 0; j < 4; ++j) {
      int mb0 = bm + wm * 32 + i * 16;
      int col = bn + wn * 64 + j * 16 + l15;
      float bia = bias[col];
      int s = 0, h = 0, d = 0;
      if (EPI == EPI_QKV) { s = col / Cc; int hd = col - s * Cc; h = hd >> 6; d = hd & 63; }
      #pragma unroll
      for (int e = 0; e < 8; ++e) {
        int row = mb0 + e + half * 8;
        float v = acc[i][j][e] + bia;
        if (EPI == EPI_QKV) {
          int b = row >> 12, n = row & 4095;
          size_t bh = (size_t)(b * Hh + h);
          if (s == 0) {        // q: fm = elu+1, layout [bh][n][d]
            float f = v > 0.f ? v + 1.f : expf(v);
            oQ[(bh * Nn + n) * Dd + d] = (__bf16)f;
          } else if (s == 1) { // k: fm, TRANSPOSED layout [bh][d][n]
            float f = v > 0.f ? v + 1.f : expf(v);
            oKT[(bh * Dd + d) * Nn + n] = (__bf16)f;
          } else {             // v: TRANSPOSED layout [bh][e][n]
            oVT[(bh * Dd + d) * Nn + n] = (__bf16)v;
          }
        } else if (EPI == EPI_PROJ) {
          size_t idx = (size_t)row * N + col;
          oF[idx] = v + xin[idx];          // + residual x
        } else if (EPI == EPI_DOWN) {
          float gl = 0.5f * v * (1.0f + erff(v * 0.70710678118f));  // exact GELU
          oB[(size_t)row * N + col] = (__bf16)gl;
        } else {  // EPI_UP
          size_t idx = (size_t)row * N + col;
          oF[idx] = v + radd[idx];         // t = r + mlp(r)
        }
      }
    }
  }
}

// ---------------------------------------------------------------------------
// k_sum[bh][d] = sum_n kT[bh][d][n]   (one wave per row of 4096 bf16)
// ---------------------------------------------------------------------------
__global__ __launch_bounds__(256) void ksum_kernel(const __bf16* __restrict__ kT,
                                                   float* __restrict__ ksum) {
  int row = blockIdx.x * 8 + (threadIdx.x >> 5);   // < 96*64
  int lane = threadIdx.x & 31;
  const __bf16* p = kT + (size_t)row * Nn;
  float s = 0.f;
  for (int i = lane * 8; i < Nn; i += 32 * 8) {
    v8bf t = *(const v8bf*)(p + i);
    #pragma unroll
    for (int j = 0; j < 8; ++j) s += (float)t[j];
  }
  #pragma unroll
  for (int o = 16; o > 0; o >>= 1) s += __shfl_xor(s, o, 32);
  if (lane == 0) ksum[row] = s;
}

// ---------------------------------------------------------------------------
// kv[bh] = k^T @ v : 64x64 with K=4096. One block per (b,h); 8 waves tile
// output 4x2 (16x32 each) over the full K loop -> deterministic, no atomics.
// Writes kv TRANSPOSED as bf16: kvT[bh][e][d].
// ---------------------------------------------------------------------------
__global__ __launch_bounds__(256) void kv_kernel(const __bf16* __restrict__ kT,
                                                 const __bf16* __restrict__ vT,
                                                 __bf16* __restrict__ kvT) {
  int bh = blockIdx.x;
  int tid = threadIdx.x, lane = tid & 31, wave = tid >> 5;
  int wm = wave & 3, wn = wave >> 2;
  const __bf16* Kt = kT + (size_t)bh * Dd * Nn;
  const __bf16* Vt = vT + (size_t)bh * Dd * Nn;
  v8f acc[2] = {};
  for (int k0 = 0; k0 < Nn; k0 += 32) {
    v16bf a = load_frag_a(Kt, Nn, wm * 16, k0, lane);
    #pragma unroll
    for (int j = 0; j < 2; ++j) {
      v16bf b = load_frag_b(Vt, Nn, wn * 32 + j * 16, k0, lane);
      acc[j] = wmma_bf16(a, b, acc[j]);
    }
  }
  int half = lane >> 4, l15 = lane & 15;
  #pragma unroll
  for (int j = 0; j < 2; ++j)
    #pragma unroll
    for (int e = 0; e < 8; ++e) {
      int d = wm * 16 + e + half * 8;
      int ec = wn * 32 + j * 16 + l15;
      kvT[(size_t)bh * Dd * Dd + (size_t)ec * Dd + d] = (__bf16)acc[j][e];
    }
}

// ---------------------------------------------------------------------------
// normalizer[bh][n] = eps + q[bh][n][:] . ksum[bh][:]
// ---------------------------------------------------------------------------
__global__ __launch_bounds__(256) void norm_kernel(const __bf16* __restrict__ qf,
                                                   const float* __restrict__ ksum,
                                                   float* __restrict__ nrm) {
  int idx = blockIdx.x * 256 + threadIdx.x;   // < 96*4096
  int bh = idx >> 12;
  const __bf16* q = qf + (size_t)idx * Dd;
  const float* ks = ksum + bh * Dd;
  float s = 1e-6f;
  #pragma unroll
  for (int d = 0; d < Dd; ++d) s += (float)q[d] * ks[d];
  nrm[idx] = s;
}

// ---------------------------------------------------------------------------
// attn = (q @ kv) / normalizer, merged heads -> bf16 [b][n][h*64+e]
// Grid: (N/128, B*H). 8 waves, each 16 rows x 64 cols, K=64.
// ---------------------------------------------------------------------------
__global__ __launch_bounds__(256) void attn_out_kernel(
    const __bf16* __restrict__ qf, const __bf16* __restrict__ kvT,
    const float* __restrict__ nrm, __bf16* __restrict__ attn) {
  int bh = blockIdx.y;
  int b = bh / Hh, h = bh - b * Hh;
  int lane = threadIdx.x & 31, wave = threadIdx.x >> 5;
  int m0 = blockIdx.x * 128 + wave * 16;
  const __bf16* Q  = qf  + (size_t)bh * Nn * Dd;
  const __bf16* KV = kvT + (size_t)bh * Dd * Dd;
  const float*  NR = nrm + (size_t)bh * Nn;
  v8f acc[4] = {};
  #pragma unroll
  for (int kk = 0; kk < Dd; kk += 32) {
    v16bf a = load_frag_a(Q, Dd, m0, kk, lane);
    #pragma unroll
    for (int j = 0; j < 4; ++j) {
      v16bf bfr = load_frag_b(KV, Dd, j * 16, kk, lane);
      acc[j] = wmma_bf16(a, bfr, acc[j]);
    }
  }
  int half = lane >> 4, l15 = lane & 15;
  #pragma unroll
  for (int j = 0; j < 4; ++j)
    #pragma unroll
    for (int e = 0; e < 8; ++e) {
      int n = m0 + e + half * 8;
      int ec = j * 16 + l15;
      float v = acc[j][e] / NR[n];
      attn[((size_t)b * Nn + n) * Cc + h * Dd + ec] = (__bf16)v;
    }
}

// ---------------------------------------------------------------------------
// Host-side orchestration
// ---------------------------------------------------------------------------
extern "C" void kernel_launch(void* const* d_in, const int* in_sizes, int n_in,
                              void* d_out, int out_size, void* d_ws, size_t ws_size,
                              hipStream_t stream) {
  const float* x      = (const float*)d_in[0];
  const float* w_qkv  = (const float*)d_in[1];
  const float* b_qkv  = (const float*)d_in[2];
  const float* w_proj = (const float*)d_in[3];
  const float* b_proj = (const float*)d_in[4];
  const float* g1     = (const float*)d_in[5];
  const float* beta1  = (const float*)d_in[6];
  const float* g2     = (const float*)d_in[7];
  const float* beta2  = (const float*)d_in[8];
  const float* w_down = (const float*)d_in[9];
  const float* b_down = (const float*)d_in[10];
  const float* w_up   = (const float*)d_in[11];
  const float* b_up   = (const float*)d_in[12];
  const float* g3     = (const float*)d_in[13];
  const float* beta3  = (const float*)d_in[14];
  float* out = (float*)d_out;

  char* ws = (char*)d_ws;
  // Workspace layout (bytes, 256B-aligned); total ~411 MB.
  __bf16* wqkvT  = (__bf16*)(ws + 0);            // 2304x768 bf16
  __bf16* wprojT = (__bf16*)(ws + 3538944);      // 768x768
  __bf16* wdownT = (__bf16*)(ws + 4718592);      // 384x768
  __bf16* wupT   = (__bf16*)(ws + 5308416);      // 768x384
  __bf16* yb     = (__bf16*)(ws + 5898240);      // LN1(x) bf16 ; later attn
  __bf16* qf     = (__bf16*)(ws + 56229888);     // q feature [bh][n][d]; later rb
  __bf16* kfT    = (__bf16*)(ws + 106561536);    // k feature [bh][d][n]; later h1
  __bf16* vfT    = (__bf16*)(ws + 156893184);    // v [bh][e][n]
  __bf16* kvbT   = (__bf16*)(ws + 207224832);    // kv^T bf16 [bh][e][d]
  float*  ksum   = (float*) (ws + 208011264);    // [bh][d]
  float*  nrm    = (float*) (ws + 208035840);    // [bh][n]
  float*  x2     = (float*) (ws + 209608704);    // x + attn-proj ; later t
  float*  r      = (float*) (ws + 310272000);    // LN2 f32
  __bf16* rb   = qf;    // LN2 bf16 (qf dead after attn_out)
  __bf16* h1   = kfT;   // GELU bottleneck bf16 (kfT dead after kv/ksum)
  __bf16* attn = yb;    // merged-head attention output (yb dead after qkv GEMM)
  float*  t    = x2;    // r + mlp(r) (x2 dead after LN2)

  // 1) Weight convert + transpose to bf16 [n][k]
  transpose_bf16_kernel<<<(Cc * 3 * Cc + 255) / 256, 256, 0, stream>>>(w_qkv, wqkvT, Cc, 3 * Cc);
  transpose_bf16_kernel<<<(Cc * Cc + 255) / 256, 256, 0, stream>>>(w_proj, wprojT, Cc, Cc);
  transpose_bf16_kernel<<<(Cc * BNd + 255) / 256, 256, 0, stream>>>(w_down, wdownT, Cc, BNd);
  transpose_bf16_kernel<<<(BNd * Cc + 255) / 256, 256, 0, stream>>>(w_up, wupT, BNd, Cc);

  // 2) LN1 -> bf16
  layernorm_kernel<<<MTOT, 256, 0, stream>>>(x, g1, beta1, nullptr, yb);

  // 3) qkv GEMM + feature map, q/kT/vT scatter
  gemm_bf16_kernel<EPI_QKV><<<dim3(MTOT / 128, (3 * Cc) / 128), 256, 0, stream>>>(
      yb, wqkvT, b_qkv, MTOT, 3 * Cc, Cc, qf, kfT, vfT, nullptr, nullptr, nullptr, nullptr);

  // 4) k_sum ; 5) kv = k^T v (WMMA) ; 6) normalizer
  ksum_kernel<<<(Bb * Hh * Dd) / 8, 256, 0, stream>>>(kfT, ksum);
  kv_kernel<<<Bb * Hh, 256, 0, stream>>>(kfT, vfT, kvbT);
  norm_kernel<<<(Bb * Hh * Nn) / 256, 256, 0, stream>>>(qf, ksum, nrm);

  // 7) out = (q @ kv)/norm, merge heads
  attn_out_kernel<<<dim3(Nn / 128, Bb * Hh), 256, 0, stream>>>(qf, kvbT, nrm, attn);

  // 8) proj GEMM + residual x -> x2 (f32)
  gemm_bf16_kernel<EPI_PROJ><<<dim3(MTOT / 128, Cc / 128), 256, 0, stream>>>(
      attn, wprojT, b_proj, MTOT, Cc, Cc, nullptr, nullptr, nullptr, x, x2, nullptr, nullptr);

  // 9) LN2 -> r (f32) + rb (bf16)
  layernorm_kernel<<<MTOT, 256, 0, stream>>>(x2, g2, beta2, r, rb);

  // 10) down GEMM + exact GELU -> h1 (bf16)
  gemm_bf16_kernel<EPI_DOWN><<<dim3(MTOT / 128, BNd / 128), 256, 0, stream>>>(
      rb, wdownT, b_down, MTOT, BNd, Cc, nullptr, nullptr, nullptr, nullptr, nullptr, h1, nullptr);

  // 11) up GEMM + residual r -> t (f32)
  gemm_bf16_kernel<EPI_UP><<<dim3(MTOT / 128, Cc / 128), 256, 0, stream>>>(
      h1, wupT, b_up, MTOT, Cc, BNd, nullptr, nullptr, nullptr, nullptr, t, nullptr, r);

  // 12) LN3 -> out (f32)
  layernorm_kernel<<<MTOT, 256, 0, stream>>>(t, g3, beta3, out, nullptr);

  (void)in_sizes; (void)n_in; (void)out_size; (void)ws_size;
}
